// PraxisAttention_8881992368482
// MI455X (gfx1250) — matrix operand order repaired and checked by reading, not compiled, not run on hardware
//
#include <hip/hip_runtime.h>
#include <hip/hip_bf16.h>
#include <math.h>

typedef __attribute__((ext_vector_type(16))) _Float16 v16h;
typedef __attribute__((ext_vector_type(8)))  _Float16 v8h;
typedef __attribute__((ext_vector_type(4)))  _Float16 v4h;
typedef __attribute__((ext_vector_type(8)))  float    v8f;

#define B_  2
#define S_  2048
#define D_  2048
#define H_  16
#define DH_ 128
#define M_  4096   // B_*S_

// ---------------------------------------------------------------------------
// CDNA5 async tile staging: GLOBAL_LOAD_ASYNC_TO_LDS_B128 copies 16B per lane
// straight from global memory into LDS (tracked by ASYNCcnt, no VGPR bounce).
// LDS aperture puts the wave-relative LDS offset in addr[31:0], so truncating
// the generic pointer yields the VDST byte offset the instruction expects.
// ---------------------------------------------------------------------------
__device__ __forceinline__ void async_copy_b128(_Float16* lds_ptr,
                                                const _Float16* gbase,
                                                unsigned gbyte_off) {
    unsigned l = (unsigned)(uintptr_t)lds_ptr;
    asm volatile("global_load_async_to_lds_b128 %0, %1, %2"
                 :: "v"(l), "v"(gbyte_off), "s"(gbase)
                 : "memory");
}

__device__ __forceinline__ void wait_async() {
    asm volatile("s_wait_asynccnt 0x0" ::: "memory");
}

// ---------------------------------------------------------------------------
// Fragment assembly: two 16-byte LDS loads -> one v16h WMMA operand.
// A-operand (16x32 f16): lane l holds row M=l&15; half-wave selects K 0-7/16-23
// vs 8-15/24-31.  B-operand (32x16 f16): lane l holds col N=l&15; half-wave
// selects K 0-7/16-23 vs 8-15/24-31, same as A.
// ---------------------------------------------------------------------------
__device__ __forceinline__ v16h make_frag(const _Float16* p, int o0, int o1) {
    v8h lo = *(const v8h*)(p + o0);
    v8h hi = *(const v8h*)(p + o1);
    return __builtin_shufflevector(lo, hi,
        0, 1, 2, 3, 4, 5, 6, 7, 8, 9, 10, 11, 12, 13, 14, 15);
}

__device__ __forceinline__ v8f wmma_f16(v16h a, v16h b, v8f c) {
    return __builtin_amdgcn_wmma_f32_16x16x32_f16(
        false, a, false, b, (short)0, c, false, false);
}

// ---------------------------------------------------------------------------
// fp32 -> f16 conversion (vectorized x4)
// ---------------------------------------------------------------------------
__global__ __launch_bounds__(256) void cvt_kernel(const float* __restrict__ src,
                                                  _Float16* __restrict__ dst,
                                                  int n) {
    int i = (blockIdx.x * 256 + threadIdx.x) * 4;
    if (i < n) {
        float4 f = *(const float4*)&src[i];
        v4h h = { (_Float16)f.x, (_Float16)f.y, (_Float16)f.z, (_Float16)f.w };
        *(v4h*)&dst[i] = h;
    }
}

// ---------------------------------------------------------------------------
// WMMA GEMM:  C[M,N] = A[M,K] * W[N,K]^T   (both row-major along K)
// Block tile 128(M) x 128(N), BK = 64.  8 waves in 4(M) x 2(N); each wave owns
// a 32x64 output tile (8 f32 accumulators, 16 WMMAs per K-stage).
// MODE 0: f16 out, remap [M,N] -> [B,H,S,Dh]   (Q, K)
// MODE 1: f16 out, remap [M,N] -> [B,H,Dh,S]   (V transposed)
// MODE 2: f32 out, plain row-major             (final projection)
// ---------------------------------------------------------------------------
template <int MODE>
__global__ __launch_bounds__(256) void gemm_kernel(const _Float16* __restrict__ A,
                                                   const _Float16* __restrict__ W,
                                                   void* __restrict__ Cout) {
    __shared__ _Float16 As[128 * 72];  // 64 f16 + 8 pad per row
    __shared__ _Float16 Bs[128 * 72];

    const int tid  = threadIdx.x;
    const int lane = tid & 31;
    const int wv   = tid >> 5;
    const int m0   = blockIdx.y * 128;
    const int n0   = blockIdx.x * 128;
    const int wm   = (wv & 3) * 32;    // wave M offset within block tile
    const int wn   = (wv >> 2) * 64;   // wave N offset within block tile

    const int nl    = lane & 15;
    const int hi    = lane >> 4;
    const int khalf = hi ? 8 : 0;      // A-operand K split
    const int koff  = hi ? 8 : 0;      // B-operand K split

    v8f acc[2][4] = {};

    for (int kt = 0; kt < D_; kt += 64) {
        __syncthreads();
        // Stage A and W tiles: 128 rows x 64 f16 = 1024 x 16B chunks each
#pragma unroll
        for (int i = 0; i < 4; ++i) {
            int cid = tid + 256 * i;
            int row = cid >> 3, ch = cid & 7;
            async_copy_b128(&As[row * 72 + ch * 8], A,
                            (unsigned)(((m0 + row) * D_ + kt + ch * 8) * 2));
            async_copy_b128(&Bs[row * 72 + ch * 8], W,
                            (unsigned)(((n0 + row) * D_ + kt + ch * 8) * 2));
        }
        wait_async();
        __syncthreads();

#pragma unroll
        for (int kk = 0; kk < 64; kk += 32) {
            v16h af0 = make_frag(&As[(wm + nl) * 72 + kk],      khalf, khalf + 16);
            v16h af1 = make_frag(&As[(wm + 16 + nl) * 72 + kk], khalf, khalf + 16);
#pragma unroll
            for (int nt = 0; nt < 4; ++nt) {
                v16h bf = make_frag(&Bs[(wn + nt * 16 + nl) * 72 + kk],
                                    koff, koff + 16);
                acc[0][nt] = wmma_f16(af0, bf, acc[0][nt]);
                acc[1][nt] = wmma_f16(af1, bf, acc[1][nt]);
            }
        }
    }

    // Epilogue: C/D layout — lane l, VGPR j -> (m = j + 8*hi, n = l&15)
#pragma unroll
    for (int mi = 0; mi < 2; ++mi) {
#pragma unroll
        for (int nt = 0; nt < 4; ++nt) {
#pragma unroll
            for (int j = 0; j < 8; ++j) {
                int m = m0 + wm + mi * 16 + j + 8 * hi;
                int n = n0 + wn + nt * 16 + nl;
                float v = acc[mi][nt][j];
                if (MODE == 2) {
                    ((float*)Cout)[(size_t)m * D_ + n] = v;
                } else {
                    int b = m >> 11, s = m & (S_ - 1);
                    int h = n >> 7, dh = n & (DH_ - 1);
                    if (MODE == 0)
                        ((_Float16*)Cout)[((size_t)((b * H_ + h) * S_ + s)) * DH_ + dh] =
                            (_Float16)v;
                    else
                        ((_Float16*)Cout)[((size_t)((b * H_ + h) * DH_ + dh)) * S_ + s] =
                            (_Float16)v;
                }
            }
        }
    }
}

// ---------------------------------------------------------------------------
// Flash attention with ALiBi + causal + key mask.
// grid = (S/128, H, B), block = 256 (8 waves).  Wave w owns q rows
// [q0+16w, q0+16w+16).  Key tiles of 32; K and V^T staged via async-to-LDS.
// ---------------------------------------------------------------------------
__global__ __launch_bounds__(256) void attn_kernel(const _Float16* __restrict__ Q,
                                                   const _Float16* __restrict__ K,
                                                   const _Float16* __restrict__ VT,
                                                   const float* __restrict__ mask,
                                                   _Float16* __restrict__ Aout) {
    __shared__ _Float16 Ks[32 * 136];      // 32 keys x 128 dh (+8 pad)
    __shared__ _Float16 Vs[128 * 40];      // 128 dh x 32 keys (+8 pad)
    __shared__ _Float16 Ps[8 * 16 * 32];   // per-wave P tile, 16 q x 32 keys

    const int tid  = threadIdx.x;
    const int lane = tid & 31;
    const int wv   = tid >> 5;
    const int b    = blockIdx.z;
    const int h    = blockIdx.y;
    const int q0   = blockIdx.x * 128;

    const int nl    = lane & 15;
    const int hi    = lane >> 4;
    const int khalf = hi ? 8 : 0;
    const int koff  = hi ? 8 : 0;

    const size_t bh = (size_t)(b * H_ + h);
    const _Float16* Qb = Q  + bh * (size_t)S_ * DH_;
    const _Float16* Kb = K  + bh * (size_t)S_ * DH_;
    const _Float16* Vb = VT + bh * (size_t)DH_ * S_;

    const float slope = exp2f(-0.5f * (float)(h + 1));   // 2^(-8(h+1)/16)
    const float scale = 0.08838834764831845f;            // 1/sqrt(128)

    // Q fragments for this wave's 16 rows, all 4 dh-chunks (held in regs)
    const int qrow = q0 + wv * 16 + nl;
    const _Float16* qp = Qb + (size_t)qrow * DH_;
    v16h qfrag[4];
#pragma unroll
    for (int c = 0; c < 4; ++c)
        qfrag[c] = make_frag(qp, c * 32 + khalf, c * 32 + khalf + 16);

    v8f oacc[8] = {};
    float m_i[8], l_i[8];
#pragma unroll
    for (int j = 0; j < 8; ++j) { m_i[j] = -__builtin_inff(); l_i[j] = 0.0f; }

    const int ktiles = (q0 + 128) >> 5;   // causal: uniform per block
    for (int t = 0; t < ktiles; ++t) {
        const int k0 = t * 32;
        __syncthreads();
        // Stage K tile (32x128) and V^T tile (128x32): 512 x 16B chunks each
#pragma unroll
        for (int i = 0; i < 2; ++i) {
            int cid = tid + 256 * i;
            int kr = cid >> 4, kc = cid & 15;
            async_copy_b128(&Ks[kr * 136 + kc * 8], Kb,
                            (unsigned)(((k0 + kr) * DH_ + kc * 8) * 2));
            int vr = cid >> 2, vc = cid & 3;
            async_copy_b128(&Vs[vr * 40 + vc * 8], Vb,
                            (unsigned)((vr * S_ + k0 + vc * 8) * 2));
        }
        wait_async();
        __syncthreads();

        // Scores: two 16x16 tiles (keys k0..+15, k0+16..+31), K-dim = 128 dh
        v8f s0 = {}, s1 = {};
#pragma unroll
        for (int c = 0; c < 4; ++c) {
            v16h kf0 = make_frag(&Ks[nl * 136],        c * 32 + koff, c * 32 + koff + 16);
            v16h kf1 = make_frag(&Ks[(16 + nl) * 136], c * 32 + koff, c * 32 + koff + 16);
            s0 = wmma_f16(qfrag[c], kf0, s0);
            s1 = wmma_f16(qfrag[c], kf1, s1);
        }

        const int key0 = k0 + nl;
        const int key1 = key0 + 16;
        const float mk0 = mask[(size_t)b * S_ + key0];
        const float mk1 = mask[(size_t)b * S_ + key1];

        float alpha[8];
#pragma unroll
        for (int j = 0; j < 8; ++j) {
            const int qr = q0 + wv * 16 + j + 8 * hi;
            float a0 = s0[j] * scale - slope * (float)(qr - key0);
            float a1 = s1[j] * scale - slope * (float)(qr - key1);
            if (key0 > qr) a0 = -__builtin_inff();
            if (key1 > qr) a1 = -__builtin_inff();
            a0 *= mk0; a1 *= mk1;
            // row max over the 16 lanes of each half-wave group
            float mt = fmaxf(a0, a1);
#pragma unroll
            for (int off = 8; off; off >>= 1)
                mt = fmaxf(mt, __shfl_xor(mt, off, 32));
            const float mn = fmaxf(m_i[j], mt);
            alpha[j] = __expf(m_i[j] - mn);
            m_i[j] = mn;
            const float p0 = __expf(a0 - mn);
            const float p1 = __expf(a1 - mn);
            s0[j] = p0; s1[j] = p1;
            float rs = p0 + p1;
#pragma unroll
            for (int off = 8; off; off >>= 1)
                rs += __shfl_xor(rs, off, 32);
            l_i[j] = l_i[j] * alpha[j] + rs;
        }

        // Rescale running output by alpha (row j <-> VGPR j, same mapping)
#pragma unroll
        for (int nt = 0; nt < 8; ++nt)
#pragma unroll
            for (int j = 0; j < 8; ++j)
                oacc[nt][j] *= alpha[j];

        // P (16q x 32key) through per-wave LDS into A-fragment layout
        _Float16* pw = &Ps[wv * 16 * 32];
#pragma unroll
        for (int j = 0; j < 8; ++j) {
            pw[(j + 8 * hi) * 32 + nl]      = (_Float16)s0[j];
            pw[(j + 8 * hi) * 32 + 16 + nl] = (_Float16)s1[j];
        }
        v16h pfrag = make_frag(&pw[nl * 32], khalf, khalf + 16);

        // O += P * V : K-dim = 32 keys, 8 dh sub-tiles of 16
#pragma unroll
        for (int nt = 0; nt < 8; ++nt) {
            v16h vf = make_frag(&Vs[(nt * 16 + nl) * 40], koff, koff + 16);
            oacc[nt] = wmma_f16(pfrag, vf, oacc[nt]);
        }
    }

    // Normalize and store as f16 [B, S, H*Dh] (input to final projection)
#pragma unroll
    for (int nt = 0; nt < 8; ++nt) {
#pragma unroll
        for (int j = 0; j < 8; ++j) {
            const int qr = q0 + wv * 16 + j + 8 * hi;
            const int dh = nt * 16 + nl;
            const float v = oacc[nt][j] / l_i[j];
            Aout[((size_t)(b * S_ + qr)) * D_ + h * DH_ + dh] = (_Float16)v;
        }
    }
}

// ---------------------------------------------------------------------------
// Host launcher
// ---------------------------------------------------------------------------
extern "C" void kernel_launch(void* const* d_in, const int* in_sizes, int n_in,
                              void* d_out, int out_size, void* d_ws, size_t ws_size,
                              hipStream_t stream) {
    const float* X    = (const float*)d_in[0];
    const float* mask = (const float*)d_in[1];
    const float* Wq   = (const float*)d_in[2];
    const float* Wk   = (const float*)d_in[3];
    const float* Wv   = (const float*)d_in[4];
    const float* Wo   = (const float*)d_in[5];

    char* ws = (char*)d_ws;
    const size_t XH  = (size_t)M_ * D_ * 2;   // 16 MB
    const size_t WH  = (size_t)D_ * D_ * 2;   //  8 MB
    _Float16* Xh  = (_Float16*)(ws);
    _Float16* Wqh = (_Float16*)(ws + XH);
    _Float16* Wkh = (_Float16*)(ws + XH + WH);
    _Float16* Wvh = (_Float16*)(ws + XH + 2 * WH);
    _Float16* Woh = (_Float16*)(ws + XH + 3 * WH);
    _Float16* Qh  = (_Float16*)(ws + XH + 4 * WH);
    _Float16* Kh  = (_Float16*)(ws + 2 * XH + 4 * WH);
    _Float16* VTh = (_Float16*)(ws + 3 * XH + 4 * WH);
    _Float16* Ah  = (_Float16*)(ws + 4 * XH + 4 * WH);

    // fp32 -> f16
    cvt_kernel<<<(M_ * D_) / 1024, 256, 0, stream>>>(X, Xh, M_ * D_);
    cvt_kernel<<<(D_ * D_) / 1024, 256, 0, stream>>>(Wq, Wqh, D_ * D_);
    cvt_kernel<<<(D_ * D_) / 1024, 256, 0, stream>>>(Wk, Wkh, D_ * D_);
    cvt_kernel<<<(D_ * D_) / 1024, 256, 0, stream>>>(Wv, Wvh, D_ * D_);
    cvt_kernel<<<(D_ * D_) / 1024, 256, 0, stream>>>(Wo, Woh, D_ * D_);

    // Projections (WMMA GEMMs), block tile 128x128
    dim3 gg(D_ / 128, M_ / 128);
    gemm_kernel<0><<<gg, 256, 0, stream>>>(Xh, Wqh, Qh);
    gemm_kernel<0><<<gg, 256, 0, stream>>>(Xh, Wkh, Kh);
    gemm_kernel<1><<<gg, 256, 0, stream>>>(Xh, Wvh, VTh);

    // Flash attention
    dim3 ga(S_ / 128, H_, B_);
    attn_kernel<<<ga, 256, 0, stream>>>(Qh, Kh, VTh, mask, Ah);

    // Output projection -> fp32 d_out
    gemm_kernel<2><<<gg, 256, 0, stream>>>(Ah, Woh, (float*)d_out);
}
